// AlignmentModel_17154099380444
// MI455X (gfx1250) — matrix operand order repaired and checked by reading, not compiled, not run on hardware
//
#include <hip/hip_runtime.h>
#include <hip/hip_bf16.h>
#include <math.h>
#include <stdint.h>

// ---------------- dims (match reference) ----------------
#define V_    30000
#define D_    300
#define H_    300
#define NS_   10000
#define L_    32
#define B_    256
#define NMAX_ 128
#define MMAX_ 128
#define ITERS_ 50
#define EPSF  0.1f
#define INVE  10.0f
#define NEGF  1e9f

// padded dims for WMMA tiling
#define KP    320        // K padded to multiple of 32 (300 -> 320)
#define NP    304        // N padded to multiple of 16 (300 -> 304)
#define MT_NS 625        // 10000/16
#define NT_H  19         // 304/16

typedef _Float16 v16h __attribute__((ext_vector_type(16)));
typedef _Float16 v8h  __attribute__((ext_vector_type(8)));
typedef float    v8f  __attribute__((ext_vector_type(8)));

// ------------- WMMA tile loaders (wave32, contiguous-k layout, same in A and B) -------------
// A: row-major M x K, f16.  Lane holds row m = M0 + (lane&15).
// VGPR0..7 hold K = koff..koff+15,
// koff = (lane>=16) ? 16 : 0.  Two contiguous 16B loads.
__device__ __forceinline__ v16h load_a16(const _Float16* A, int lda, int m0, int k0, int lane) {
  int m = m0 + (lane & 15);
  int ko = (lane & 16) ? 16 : 0;
  const _Float16* p = A + (size_t)m * lda + k0 + ko;
  v8h lo = *(const v8h*)(p);
  v8h hi = *(const v8h*)(p + 8);
  v16h r;
#pragma unroll
  for (int i = 0; i < 8; ++i) { r[i] = lo[i]; r[i + 8] = hi[i]; }
  return r;
}

// B stored N-major ([N][Kpad], i.e. pre-transposed weight).  Lane holds
// column n = N0 + (lane&15); K = k0 + (lane>=16?16:0) + i, 16 contiguous halves.
__device__ __forceinline__ v16h load_b_nmajor(const _Float16* Bt, int ldb, int n0, int k0, int lane) {
  int n  = n0 + (lane & 15);
  int kk = k0 + ((lane & 16) ? 16 : 0);
  const _Float16* p = Bt + (size_t)n * ldb + kk;
  v8h lo = *(const v8h*)(p);
  v8h hi = *(const v8h*)(p + 8);
  v16h r;
#pragma unroll
  for (int i = 0; i < 8; ++i) { r[i] = lo[i]; r[i + 8] = hi[i]; }
  return r;
}

// B stored row-major ([K][N]); strided per-element gather.
__device__ __forceinline__ v16h load_b_rowmajor(const _Float16* B, int ldb, int k0, int n0, int lane) {
  int n  = n0 + (lane & 15);
  int kb = k0 + ((lane & 16) ? 16 : 0);
  v16h r;
#pragma unroll
  for (int i = 0; i < 16; ++i) r[i] = B[(size_t)(kb + i) * ldb + n];
  return r;
}

#define WMMA_F16(a, b, c) \
  __builtin_amdgcn_wmma_f32_16x16x32_f16(false, (a), false, (b), (short)0, (c), false, false)

// ---------------- 1) weight pack: f32 [K][N] -> f16 N-major [NP][KP] ----------------
__global__ void pack_wT_kernel(const float* __restrict__ W, _Float16* __restrict__ Wt,
                               int Kv, int Nv, int ld, int rowOff) {
  int n = blockIdx.x;      // 0..NP-1
  int k = threadIdx.x;     // 0..KP-1
  float v = (k < Kv && n < Nv) ? W[(size_t)(rowOff + k) * ld + n] : 0.0f;
  Wt[(size_t)n * KP + k] = (_Float16)v;
}

// ---------------- 2) embedding gather + masked mean pool ----------------
__global__ void embed_pool_kernel(const int* __restrict__ data, const float* __restrict__ emb,
                                  _Float16* __restrict__ enc) {
  int s = blockIdx.x;        // sample
  int tid = threadIdx.x;     // 128 threads
  const int* row = data + (size_t)s * L_;
  float a0 = 0.f, a1 = 0.f, a2 = 0.f;
  int cnt = 0;
  for (int l = 0; l < L_; ++l) {
    int tok = row[l];
    if (tok != 0) {           // PAD == 0
      ++cnt;
      const float* e = emb + (size_t)tok * D_;
      a0 += e[tid];
      a1 += e[tid + 128];
      if (tid + 256 < D_) a2 += e[tid + 256];
    }
  }
  float inv = 1.0f / (float)(cnt > 0 ? cnt : 1);
  _Float16* dst = enc + (size_t)s * KP;
  dst[tid]       = (_Float16)(a0 * inv);
  dst[tid + 128] = (_Float16)(a1 * inv);
  int d2 = tid + 256;
  if (d2 < KP) dst[d2] = (_Float16)((d2 < D_) ? a2 * inv : 0.0f);
}

// ---------------- 3) FFN GEMM: out = relu(A @ Wt^T + bias) ----------------
// A: [M][KP] f16.  Wt: [NP][KP] f16 (N-major).  Writes f16 (ldo=KP) or f32 (ldo=NP).
__global__ void ffn_gemm_kernel(const _Float16* __restrict__ A, const _Float16* __restrict__ Wt,
                                const float* __restrict__ bias,
                                _Float16* __restrict__ outH, float* __restrict__ outF,
                                int mtiles, int ldo) {
  int lane = threadIdx.x;
  int tile = blockIdx.x * blockDim.y + threadIdx.y;
  if (tile >= mtiles * NT_H) return;            // wave-uniform exit
  int m0 = (tile / NT_H) * 16;
  int n0 = (tile % NT_H) * 16;
  v8f acc = {};
#pragma unroll 2
  for (int k0 = 0; k0 < KP; k0 += 32) {
    if (k0 + 32 < KP)
      __builtin_prefetch(A + (size_t)(m0 + (lane & 15)) * KP + k0 + 32, 0, 1);
    v16h a = load_a16(A, KP, m0, k0, lane);
    v16h b = load_b_nmajor(Wt, KP, n0, k0, lane);
    acc = WMMA_F16(a, b, acc);
  }
  int n = n0 + (lane & 15);
  float bv = (n < H_) ? bias[n] : 0.0f;
#pragma unroll
  for (int j = 0; j < 8; ++j) {
    int m = m0 + j + ((lane & 16) ? 8 : 0);
    float v = acc[j] + bv;
    v = v > 0.0f ? v : 0.0f;
    if (n >= H_) v = 0.0f;
    if (outH) outH[(size_t)m * ldo + n] = (_Float16)v;
    else      outF[(size_t)m * ldo + n] = v;
  }
}

// ---------------- 4) gather rows/cols, mask, L2-normalize ----------------
__global__ void gather_norm_kernel(const float* __restrict__ henc,
                                   const int* __restrict__ row_idx, const int* __restrict__ col_idx,
                                   const int* __restrict__ row_len, const int* __restrict__ col_len,
                                   _Float16* __restrict__ rows, _Float16* __restrict__ rn,
                                   _Float16* __restrict__ cols, _Float16* __restrict__ cn) {
  int bx = blockIdx.x;
  int b = bx >> 8;
  int i = bx & 255;
  int tid = threadIdx.x;              // 128 threads
  bool isRow = (i < NMAX_);
  int n   = isRow ? i : (i - NMAX_);
  int s   = isRow ? row_idx[(size_t)b * NMAX_ + n] : col_idx[(size_t)b * MMAX_ + n];
  int len = isRow ? row_len[b] : col_len[b];
  bool valid = n < len;
  const float* src = henc + (size_t)s * NP;
  float v0 = valid ? src[tid] : 0.0f;
  float v1 = valid ? src[tid + 128] : 0.0f;
  float v2 = (valid && (tid + 256) < H_) ? src[tid + 256] : 0.0f;
  __shared__ float red[128];
  red[tid] = v0 * v0 + v1 * v1 + v2 * v2;
  __syncthreads();
  for (int off = 64; off > 0; off >>= 1) {
    if (tid < off) red[tid] += red[tid + off];
    __syncthreads();
  }
  float inv = 1.0f / (sqrtf(red[0]) + 1e-8f);
  _Float16* dX = (isRow ? rows : cols) + ((size_t)b * NMAX_ + n) * KP;
  _Float16* dN = (isRow ? rn   : cn  ) + ((size_t)b * NMAX_ + n) * KP;
  dX[tid]       = (_Float16)v0;  dN[tid]       = (_Float16)(v0 * inv);
  dX[tid + 128] = (_Float16)v1;  dN[tid + 128] = (_Float16)(v1 * inv);
  int d2 = tid + 256;
  if (d2 < KP) {
    float vv = (d2 < H_) ? v2 : 0.0f;
    dX[d2] = (_Float16)vv; dN[d2] = (_Float16)(vv * inv);
  }
}

// ---------------- 5) batched cosine cost: Cm = masked(1 - rn @ cn^T) ----------------
__global__ void cost_kernel(const _Float16* __restrict__ rn, const _Float16* __restrict__ cn,
                            const int* __restrict__ rlenA, const int* __restrict__ clenA,
                            float* __restrict__ Cm) {
  int lane = threadIdx.x;
  int gt = blockIdx.x * blockDim.y + threadIdx.y;
  if (gt >= B_ * 64) return;
  int b = gt >> 6;
  int t = gt & 63;
  int m0 = (t >> 3) * 16;   // row (n) tile
  int n0 = (t & 7) * 16;    // col (m) tile
  const _Float16* A  = rn + (size_t)b * NMAX_ * KP;
  const _Float16* Bt = cn + (size_t)b * MMAX_ * KP;   // cn row-major == N-major B
  v8f acc = {};
  for (int k0 = 0; k0 < KP; k0 += 32) {
    v16h a = load_a16(A, KP, m0, k0, lane);
    v16h bm = load_b_nmajor(Bt, KP, n0, k0, lane);
    acc = WMMA_F16(a, bm, acc);
  }
  int rlen = rlenA[b], clen = clenA[b];
  int mc = n0 + (lane & 15);
#pragma unroll
  for (int j = 0; j < 8; ++j) {
    int nr = m0 + j + ((lane & 16) ? 8 : 0);
    float cost = 1.0f - acc[j];
    float v = (nr < rlen && mc < clen) ? cost : 1e4f;
    Cm[((size_t)b * NMAX_ + nr) * MMAX_ + mc] = v;
  }
}

// ---------------- 6) log-domain Sinkhorn, LDS-resident per batch ----------------
// The 64KB cost tile is staged into LDS with GLOBAL_LOAD_ASYNC_TO_LDS_B128
// (CDNA5 async copy, VGPR-bypassing, tracked by ASYNCcnt), fenced with
// s_wait_asynccnt before the workgroup barrier.
__global__ void __launch_bounds__(128) sinkhorn_kernel(const float* __restrict__ CmG,
                                                       const int* __restrict__ rlenA,
                                                       const int* __restrict__ clenA,
                                                       _Float16* __restrict__ P,
                                                       _Float16* __restrict__ PT) {
  __shared__ __align__(16) float sC[NMAX_ * MMAX_];   // 64 KB of 320 KB WGP LDS
  __shared__ float sf[NMAX_];
  __shared__ float sg[MMAX_];
  int b = blockIdx.x;
  int tid = threadIdx.x;
  const float* src = CmG + (size_t)b * NMAX_ * MMAX_;
  // 64KB tile: 128 lanes x 32 ops x 16B, each op copies one 16B chunk to LDS.
#pragma unroll 4
  for (int i = 0; i < 32; ++i) {
    int e = (i * 128 + tid) * 4;                         // float index of 16B chunk
    uint32_t ldsa = (uint32_t)(uintptr_t)(&sC[e]);       // flat->LDS: addr[31:0]
    const float* g = src + e;
    asm volatile("global_load_async_to_lds_b128 %0, %1, off"
                 :: "v"(ldsa), "v"(g)
                 : "memory");
  }
  asm volatile("s_wait_asynccnt 0x0" ::: "memory");
  int rlen = rlenA[b], clen = clenA[b];
  float loga = (tid < rlen) ? -logf((float)rlen) : -NEGF;
  float logb = (tid < clen) ? -logf((float)clen) : -NEGF;
  sf[tid] = 0.0f;
  sg[tid] = 0.0f;
  __syncthreads();
  for (int it = 0; it < ITERS_; ++it) {
    // f-update: thread owns row n = tid
    const float* cr = sC + (size_t)tid * MMAX_;
    float mx = -3.4e38f;
    for (int m = 0; m < MMAX_; ++m) mx = fmaxf(mx, (sg[m] - cr[m]) * INVE);
    float s = 0.0f;
    for (int m = 0; m < MMAX_; ++m) s += expf((sg[m] - cr[m]) * INVE - mx);
    float fnew = EPSF * loga - EPSF * (mx + logf(s));
    __syncthreads();
    sf[tid] = fnew;
    __syncthreads();
    // g-update: thread owns col m = tid
    float mx2 = -3.4e38f;
    for (int n = 0; n < NMAX_; ++n) mx2 = fmaxf(mx2, (sf[n] - sC[(size_t)n * MMAX_ + tid]) * INVE);
    float s2 = 0.0f;
    for (int n = 0; n < NMAX_; ++n) s2 += expf((sf[n] - sC[(size_t)n * MMAX_ + tid]) * INVE - mx2);
    float gnew = EPSF * logb - EPSF * (mx2 + logf(s2));
    __syncthreads();
    sg[tid] = gnew;
    __syncthreads();
  }
  float fv = sf[tid];
  for (int m = 0; m < MMAX_; ++m) {
    float v = 0.0f;
    if (tid < rlen && m < clen)
      v = expf((fv + sg[m] - sC[(size_t)tid * MMAX_ + m]) * INVE);
    _Float16 h = (_Float16)v;
    P [((size_t)b * NMAX_ + tid) * MMAX_ + m]   = h;
    PT[((size_t)b * MMAX_ + m)   * NMAX_ + tid] = h;
  }
}

// ---------------- 7) attend: out = Pmat @ X  (per batch) ----------------
// Pmat: [128][128] f16 (A), X: [128][KP] f16 row-major (B), out: [128][KP] f16
__global__ void attend_kernel(const _Float16* __restrict__ Pmat, const _Float16* __restrict__ X,
                              _Float16* __restrict__ out) {
  int lane = threadIdx.x;
  int gt = blockIdx.x * blockDim.y + threadIdx.y;
  if (gt >= B_ * 8 * NT_H) return;
  int b = gt / (8 * NT_H);
  int t = gt % (8 * NT_H);
  int m0 = (t / NT_H) * 16;
  int n0 = (t % NT_H) * 16;
  const _Float16* A = Pmat + (size_t)b * NMAX_ * MMAX_;
  const _Float16* Bm = X   + (size_t)b * NMAX_ * KP;
  v8f acc = {};
  for (int k0 = 0; k0 < 128; k0 += 32) {
    v16h a  = load_a16(A, 128, m0, k0, lane);
    v16h bm = load_b_rowmajor(Bm, KP, k0, n0, lane);
    acc = WMMA_F16(a, bm, acc);
  }
  int n = n0 + (lane & 15);
  _Float16* dst = out + (size_t)b * NMAX_ * KP;
#pragma unroll
  for (int j = 0; j < 8; ++j) {
    int m = m0 + j + ((lane & 16) ? 8 : 0);
    dst[(size_t)m * KP + n] = (_Float16)acc[j];
  }
}

// ---------------- 8) compare GEMM + fused masked row-sum ----------------
// acc = X1 @ Wt1 + X2 @ Wt2 (K=320 each, concat over 600), relu(+bias),
// mask rows n<len, reduce over the tile's 16 rows, atomic-add into outSum[b][h].
__global__ void compare_kernel(const _Float16* __restrict__ X1, const _Float16* __restrict__ X2,
                               const _Float16* __restrict__ Wt1, const _Float16* __restrict__ Wt2,
                               const float* __restrict__ bias, const int* __restrict__ lens,
                               float* __restrict__ outSum) {
  int lane = threadIdx.x;
  int tile = blockIdx.x * blockDim.y + threadIdx.y;
  if (tile >= 2048 * NT_H) return;     // M = B*128 = 32768 -> 2048 m-tiles
  int m0 = (tile / NT_H) * 16;
  int n0 = (tile % NT_H) * 16;
  v8f acc = {};
  for (int k0 = 0; k0 < KP; k0 += 32) {
    v16h a1 = load_a16(X1, KP, m0, k0, lane);
    v16h b1 = load_b_nmajor(Wt1, KP, n0, k0, lane);
    acc = WMMA_F16(a1, b1, acc);
    v16h a2 = load_a16(X2, KP, m0, k0, lane);
    v16h b2 = load_b_nmajor(Wt2, KP, n0, k0, lane);
    acc = WMMA_F16(a2, b2, acc);
  }
  int n = n0 + (lane & 15);
  float bv = (n < H_) ? bias[n] : 0.0f;
  int b = m0 >> 7;                     // 128 rows per batch element; tile within one b
  int len = lens[b];
  float partial = 0.0f;
#pragma unroll
  for (int j = 0; j < 8; ++j) {
    int m = m0 + j + ((lane & 16) ? 8 : 0);
    int nr = m & 127;
    float v = acc[j] + bv;
    v = v > 0.0f ? v : 0.0f;
    if (n >= H_) v = 0.0f;
    if (nr < len) partial += v;
  }
  partial += __shfl_xor(partial, 16, 32);    // combine M halves (lane L with L+16)
  if (lane < 16 && n < H_)
    atomicAdd(&outSum[(size_t)b * NP + n], partial);
}

// ---------------- 9) classifier head (tiny; scalar) ----------------
__global__ void cls_kernel(const float* __restrict__ cr, const float* __restrict__ cc,
                           const float* __restrict__ Wcls, const float* __restrict__ bcls,
                           float* __restrict__ z) {
  int b = blockIdx.x;
  int h = threadIdx.x;
  if (h >= H_) return;
  float acc = bcls[h];
  for (int k = 0; k < H_; ++k) {
    acc += cr[(size_t)b * NP + k] * Wcls[(size_t)k * H_ + h];
    acc += cc[(size_t)b * NP + k] * Wcls[(size_t)(H_ + k) * H_ + h];
  }
  z[(size_t)b * H_ + h] = fmaxf(acc, 0.0f);
}

__global__ void out_kernel(const float* __restrict__ z, const float* __restrict__ Wout,
                           const float* __restrict__ bout, float* __restrict__ out) {
  int i = blockIdx.x * blockDim.x + threadIdx.x;
  if (i >= B_ * 2) return;
  int b = i >> 1, j = i & 1;
  float acc = bout[j];
  for (int k = 0; k < H_; ++k) acc += z[(size_t)b * H_ + k] * Wout[(size_t)k * 2 + j];
  out[i] = acc;
}

// ---------------- host orchestration ----------------
extern "C" void kernel_launch(void* const* d_in, const int* in_sizes, int n_in,
                              void* d_out, int out_size, void* d_ws, size_t ws_size,
                              hipStream_t stream) {
  const int*   data    = (const int*)  d_in[0];
  const int*   row_idx = (const int*)  d_in[1];
  const int*   col_idx = (const int*)  d_in[2];
  const int*   row_len = (const int*)  d_in[3];
  const int*   col_len = (const int*)  d_in[4];
  const float* emb     = (const float*)d_in[5];
  const float* W1      = (const float*)d_in[6];
  const float* b1      = (const float*)d_in[7];
  const float* W2      = (const float*)d_in[8];
  const float* b2      = (const float*)d_in[9];
  const float* Wc      = (const float*)d_in[10];
  const float* bc      = (const float*)d_in[11];
  const float* Wcls    = (const float*)d_in[12];
  const float* bcls    = (const float*)d_in[13];
  const float* Wout    = (const float*)d_in[14];
  const float* bout    = (const float*)d_in[15];
  float* out = (float*)d_out;

  // carve workspace (256B aligned)
  size_t off = 0;
  auto carve = [&](size_t bytes) -> char* {
    char* p = (char*)d_ws + off;
    off = (off + bytes + 255) & ~(size_t)255;
    return p;
  };
  _Float16* encF  = (_Float16*)carve((size_t)NS_ * KP * 2);
  _Float16* h1F   = (_Float16*)carve((size_t)NS_ * KP * 2);
  float*    hencF = (float*)   carve((size_t)NS_ * NP * 4);
  _Float16* W1T   = (_Float16*)carve((size_t)NP * KP * 2);
  _Float16* W2T   = (_Float16*)carve((size_t)NP * KP * 2);
  _Float16* WcTT  = (_Float16*)carve((size_t)NP * KP * 2);
  _Float16* WcBT  = (_Float16*)carve((size_t)NP * KP * 2);
  _Float16* rows  = (_Float16*)carve((size_t)B_ * NMAX_ * KP * 2);
  _Float16* cols  = (_Float16*)carve((size_t)B_ * MMAX_ * KP * 2);
  _Float16* rnF   = (_Float16*)carve((size_t)B_ * NMAX_ * KP * 2);
  _Float16* cnF   = (_Float16*)carve((size_t)B_ * MMAX_ * KP * 2);
  float*    CmF   = (float*)   carve((size_t)B_ * NMAX_ * MMAX_ * 4);
  _Float16* Pf    = (_Float16*)carve((size_t)B_ * NMAX_ * MMAX_ * 2);
  _Float16* PTf   = (_Float16*)carve((size_t)B_ * NMAX_ * MMAX_ * 2);
  _Float16* attC  = (_Float16*)carve((size_t)B_ * NMAX_ * KP * 2);
  _Float16* attR  = (_Float16*)carve((size_t)B_ * MMAX_ * KP * 2);
  float*    crF   = (float*)   carve((size_t)B_ * NP * 4);
  float*    ccF   = (float*)   carve((size_t)B_ * NP * 4);
  float*    zF    = (float*)   carve((size_t)B_ * H_ * 4);
  (void)ws_size; (void)in_sizes; (void)n_in; (void)out_size;

  // zero buffers whose pad regions / accumulators must start at 0
  hipMemsetAsync(h1F,  0, (size_t)NS_ * KP * 2, stream);
  hipMemsetAsync(attC, 0, (size_t)B_ * NMAX_ * KP * 2, stream);
  hipMemsetAsync(attR, 0, (size_t)B_ * MMAX_ * KP * 2, stream);
  hipMemsetAsync(crF,  0, (size_t)B_ * NP * 4, stream);
  hipMemsetAsync(ccF,  0, (size_t)B_ * NP * 4, stream);

  dim3 wblk(32, 8);   // 8 waves per block, one 16x16 tile per wave

  // pack weights -> f16 N-major
  pack_wT_kernel<<<NP, KP, 0, stream>>>(W1, W1T, D_, H_, H_, 0);
  pack_wT_kernel<<<NP, KP, 0, stream>>>(W2, W2T, H_, H_, H_, 0);
  pack_wT_kernel<<<NP, KP, 0, stream>>>(Wc, WcTT, H_, H_, H_, 0);
  pack_wT_kernel<<<NP, KP, 0, stream>>>(Wc, WcBT, H_, H_, H_, H_);

  // embed + pool
  embed_pool_kernel<<<NS_, 128, 0, stream>>>(data, emb, encF);

  // FFN: h1 = relu(enc@W1+b1) ; henc = relu(h1@W2+b2)
  int ffnTiles = MT_NS * NT_H;
  int ffnBlocks = (ffnTiles + 7) / 8;
  ffn_gemm_kernel<<<ffnBlocks, wblk, 0, stream>>>(encF, W1T, b1, h1F, nullptr, MT_NS, KP);
  ffn_gemm_kernel<<<ffnBlocks, wblk, 0, stream>>>(h1F, W2T, b2, nullptr, hencF, MT_NS, NP);

  // gather + mask + normalize
  gather_norm_kernel<<<B_ * 256, 128, 0, stream>>>(hencF, row_idx, col_idx, row_len, col_len,
                                                   rows, rnF, cols, cnF);

  // batched cosine cost
  cost_kernel<<<(B_ * 64) / 8, wblk, 0, stream>>>(rnF, cnF, row_len, col_len, CmF);

  // Sinkhorn (one workgroup per batch element, LDS-resident, async-staged)
  sinkhorn_kernel<<<B_, 128, 0, stream>>>(CmF, row_len, col_len, Pf, PTf);

  // attend: attended_col = P @ cols ; attended_row = P^T @ rows
  int attTiles = B_ * 8 * NT_H;
  attend_kernel<<<attTiles / 8, wblk, 0, stream>>>(Pf,  cols, attC);
  attend_kernel<<<attTiles / 8, wblk, 0, stream>>>(PTf, rows, attR);

  // compare + fused masked sum
  int cmpTiles = 2048 * NT_H;
  compare_kernel<<<cmpTiles / 8, wblk, 0, stream>>>(rows, attC, WcTT, WcBT, bc, row_len, crF);
  compare_kernel<<<cmpTiles / 8, wblk, 0, stream>>>(cols, attR, WcTT, WcBT, bc, col_len, ccF);

  // classifier head
  cls_kernel<<<B_, 320, 0, stream>>>(crF, ccF, Wcls, bcls, zF);
  out_kernel<<<2, 256, 0, stream>>>(zF, Wout, bout, out);
}
